// CausalSelfAttention_28570122453821
// MI455X (gfx1250) — matrix operand order repaired and checked by reading, not compiled, not run on hardware
//
#include <hip/hip_runtime.h>
#include <hip/hip_bf16.h>
#include <math.h>

// gfx1250 async global->LDS path (ASYNCcnt) via inline asm — verified to
// assemble in round 2.
#define USE_ASYNC_LDS 1

// ---------------------------------------------------------------------------
// CDNA5 (gfx1250) WMMA helpers
// ---------------------------------------------------------------------------
typedef __bf16 bf16_t;
typedef bf16_t v16bf __attribute__((ext_vector_type(16)));
typedef float  v8f   __attribute__((ext_vector_type(8)));

__device__ __forceinline__ v8f zero_v8f() {
  v8f z;
#pragma unroll
  for (int i = 0; i < 8; ++i) z[i] = 0.0f;
  return z;
}

// Native f32 -> bf16 (RNE) via hardware conversion.
__device__ __forceinline__ unsigned short f32_to_bf16(float f) {
  union { __bf16 h; unsigned short u; } c;
  c.h = (__bf16)f;
  return c.u;
}

// LDS byte offset of a __shared__ object: generic pointers to LDS carry the
// LDS byte offset in their low 32 bits (LDS aperture layout).
__device__ __forceinline__ unsigned lds_offset_of(const void* p) {
  return (unsigned)(unsigned long long)p;
}

// Async copy 16B global -> LDS (per active lane). Tracked by ASYNCcnt.
__device__ __forceinline__ void async_load_b128(unsigned lds_off,
                                                const void* gaddr) {
#if USE_ASYNC_LDS
  asm volatile("global_load_async_to_lds_b128 %0, %1, off"
               :: "v"(lds_off), "v"(gaddr) : "memory");
#else
  (void)lds_off; (void)gaddr;
#endif
}

__device__ __forceinline__ void wait_async_all() {
#if USE_ASYNC_LDS
  asm volatile("s_wait_asynccnt 0x0" ::: "memory");
#endif
}

// Load a 16-element bf16 fragment as two 16B chunks from LDS (both 16B aligned).
__device__ __forceinline__ v16bf load_frag(const unsigned short* p0,
                                           const unsigned short* p1) {
  union { uint4 q[2]; v16bf v; } u;
  u.q[0] = *(const uint4*)p0;
  u.q[1] = *(const uint4*)p1;
  return u.v;
}

__device__ __forceinline__ v8f wmma_bf16(v16bf a, v16bf b, v8f c) {
  // D = A(16x32 bf16) * B(32x16 bf16) + C(16x16 f32)
  return __builtin_amdgcn_wmma_f32_16x16x32_bf16(false, a, false, b,
                                                 (short)0, c, false, false);
}

// ---------------------------------------------------------------------------
// f32 -> bf16 conversion (vectorized, grid-stride)
// ---------------------------------------------------------------------------
__global__ void convert_f32_bf16_kernel(const float* __restrict__ src,
                                        unsigned short* __restrict__ dst,
                                        long long n4) {
  long long i = (long long)blockIdx.x * blockDim.x + threadIdx.x;
  long long stride = (long long)gridDim.x * blockDim.x;
  for (; i < n4; i += stride) {
    float4 v = ((const float4*)src)[i];
    unsigned short o[4] = { f32_to_bf16(v.x), f32_to_bf16(v.y),
                            f32_to_bf16(v.z), f32_to_bf16(v.w) };
    ((uint2*)dst)[i] = *(const uint2*)o;
  }
}

// ---------------------------------------------------------------------------
// GEMM: C[MxN] (f32) = A[MxK] (bf16, row-major) * B[KxN] (bf16, row-major)
// M%128==0, N%128==0, K%32==0. Block: 256 thr (8 waves), tile 128x128.
// Wave w: rows (w>>2)*64, cols (w&3)*32 -> 4x2 WMMA accumulators.
// ---------------------------------------------------------------------------
#define GPAD 8  // row stride 40 halfwords (80B, 16B aligned)
__global__ __launch_bounds__(256, 1)
void gemm_bf16_kernel(const unsigned short* __restrict__ A,
                      const unsigned short* __restrict__ B,
                      float* __restrict__ C, int M, int N, int K) {
  __shared__ unsigned short As [128][32 + GPAD];
  __shared__ unsigned short Bts[128][32 + GPAD];   // B transposed: [n][k]

  const int tid  = threadIdx.x;
  const int wave = tid >> 5;
  const int lane = tid & 31;
  const int lh   = lane & 15;
  const int half = lane >> 4;
  const int m0   = blockIdx.y * 128;
  const int n0   = blockIdx.x * 128;
  const int wm   = (wave >> 2) * 64;
  const int wn   = (wave & 3) * 32;

  // per-thread staging coordinates (fixed across K steps)
  const int arow0 = (tid * 2) >> 2;            // A chunk 0 row
  const int ac80  = ((tid * 2) & 3) * 8;
  const int arow1 = (tid * 2 + 1) >> 2;        // A chunk 1 row
  const int ac81  = ((tid * 2 + 1) & 3) * 8;
  const int bkp   = (tid >> 4) * 2;            // B row pair (0,2,..,30)
  const int bn8   = (tid & 15) * 8;

  const unsigned a_l0 = lds_offset_of(&As[arow0][ac80]);
  const unsigned a_l1 = lds_offset_of(&As[arow1][ac81]);

  // strength-reduced global pointers (stepped each K iteration)
  const unsigned short* ap0 = A + (size_t)(m0 + arow0) * K + ac80;
  const unsigned short* ap1 = A + (size_t)(m0 + arow1) * K + ac81;
  const unsigned short* bp0 = B + (size_t)bkp * N + n0 + bn8;
  const size_t bstep = (size_t)32 * N;

  v8f acc[4][2];
#pragma unroll
  for (int i = 0; i < 4; ++i)
#pragma unroll
    for (int j = 0; j < 2; ++j) acc[i][j] = zero_v8f();

  for (int kb = 0; kb < K; kb += 32) {
    // --- A tile: async global -> LDS, 2 x 16B per thread
    async_load_b128(a_l0, ap0);
    async_load_b128(a_l1, ap1);

    // --- B tile: 2 adjacent K rows per thread, transpose via v_perm_b32
    {
      uint4 d0 = *(const uint4*)bp0;         // row bkp
      uint4 d1 = *(const uint4*)(bp0 + N);   // row bkp+1
      const unsigned* w0 = (const unsigned*)&d0;
      const unsigned* w1 = (const unsigned*)&d1;
#pragma unroll
      for (int wd = 0; wd < 4; ++wd) {
        unsigned lo = __builtin_amdgcn_perm(w1[wd], w0[wd], 0x05040100u);
        unsigned hi = __builtin_amdgcn_perm(w1[wd], w0[wd], 0x07060302u);
        *(unsigned*)(&Bts[bn8 + 2 * wd    ][bkp]) = lo;   // 4B aligned
        *(unsigned*)(&Bts[bn8 + 2 * wd + 1][bkp]) = hi;
      }
    }
    // prefetch next K-step tiles into cache
    if (kb + 32 < K) {
      __builtin_prefetch(ap0 + 32, 0, 1);
      __builtin_prefetch(bp0 + bstep, 0, 1);
    }
    ap0 += 32; ap1 += 32; bp0 += bstep;
    wait_async_all();
    __syncthreads();

    v16bf af[4], bfr[2];
#pragma unroll
    for (int i = 0; i < 4; ++i) {
      const unsigned short* rp = &As[wm + i * 16 + lh][0];
      af[i] = load_frag(rp + half * 8, rp + 16 + half * 8);
    }
#pragma unroll
    for (int j = 0; j < 2; ++j) {
      const unsigned short* rp = &Bts[wn + j * 16 + lh][0];
      bfr[j] = load_frag(rp + half * 8, rp + 16 + half * 8);
    }
#pragma unroll
    for (int i = 0; i < 4; ++i)
#pragma unroll
      for (int j = 0; j < 2; ++j)
        acc[i][j] = wmma_bf16(af[i], bfr[j], acc[i][j]);
    __syncthreads();
  }

  // Epilogue: C/D layout lane = n + (m>=8)*16, vgpr r holds m = r + half*8.
  // Incremental pointer walk -> no per-store 64-bit multiplies.
  const size_t Ns = (size_t)N;
  float* cbase = C + (size_t)(m0 + wm + half * 8) * Ns + (n0 + wn + lh);
#pragma unroll
  for (int i = 0; i < 4; ++i) {
    float* p = cbase + (size_t)(i * 16) * Ns;
#pragma unroll
    for (int r = 0; r < 8; ++r) {
      p[0]  = acc[i][0][r];
      p[16] = acc[i][1][r];
      p += Ns;
    }
  }
}

// ---------------------------------------------------------------------------
// Fused per-head RMSNorm + RoPE, f32 in -> bf16 out. One wave per (row, head).
// src/dst layout: [rows][n_heads*128]. Position for RoPE = row % Tlen.
// ---------------------------------------------------------------------------
__global__ __launch_bounds__(256, 1)
void rmsnorm_rope_kernel(const float* __restrict__ src,
                         unsigned short* __restrict__ dst,
                         const float* __restrict__ w,
                         int n_heads, int rows, int Tlen) {
  const int g    = blockIdx.x * 8 + (threadIdx.x >> 5);
  const int lane = threadIdx.x & 31;
  const int row  = g / n_heads;
  const int h    = g - row * n_heads;
  if (row >= rows) return;
  const int t = row % Tlen;

  const size_t base = (size_t)row * (n_heads * 128) + h * 128 + lane * 4;
  float4 xv = *(const float4*)(src + base);

  float ss = xv.x * xv.x + xv.y * xv.y + xv.z * xv.z + xv.w * xv.w;
#pragma unroll
  for (int m = 16; m >= 1; m >>= 1) ss += __shfl_xor(ss, m, 32);
  const float rms = rsqrtf(ss * (1.0f / 128.0f) + 1e-5f);

  const int d = lane * 4;
  float x0 = xv.x * rms * w[d + 0];
  float x1 = xv.y * rms * w[d + 1];
  float x2 = xv.z * rms * w[d + 2];
  float x3 = xv.w * rms * w[d + 3];

  // pair indices i = d/2 and d/2+1; inv_freq = 10000^(-(2i)/128)
  const float L2B = 13.287712379549449f;  // log2(10000)
  float i0 = (float)(d >> 1);
  float i1 = i0 + 1.0f;
  float f0 = exp2f(-(2.0f * i0 / 128.0f) * L2B);
  float f1 = exp2f(-(2.0f * i1 / 128.0f) * L2B);
  float th0 = (float)t * f0, th1 = (float)t * f1;
  float c0 = cosf(th0), s0 = sinf(th0);
  float c1 = cosf(th1), s1 = sinf(th1);

  float y0 = x0 * c0 - x1 * s0;
  float y1 = x0 * s0 + x1 * c0;
  float y2 = x2 * c1 - x3 * s1;
  float y3 = x2 * s1 + x3 * c1;

  unsigned short o[4] = { f32_to_bf16(y0), f32_to_bf16(y1),
                          f32_to_bf16(y2), f32_to_bf16(y3) };
  *(uint2*)(dst + base) = *(const uint2*)o;
}

// ---------------------------------------------------------------------------
// Causal flash attention (GQA 16 q-heads / 4 kv-heads, D=128, T=2048, B=2).
// Grid: (T/128, H, B). Block: 256 thr (8 waves); wave owns 16 q rows.
// Q: [B*T][2048] bf16, K/V: [B*T][512] bf16, Y: [B*T][2048] bf16.
// ---------------------------------------------------------------------------
__global__ __launch_bounds__(256, 1)
void attention_kernel(const unsigned short* __restrict__ Q,
                      const unsigned short* __restrict__ Km,
                      const unsigned short* __restrict__ Vm,
                      unsigned short* __restrict__ Y) {
  __shared__ unsigned short Qs [128][136];      // q rows x D (pad 8)
  __shared__ unsigned short Ks [32][136];       // key rows x D
  __shared__ unsigned short Vts[128][40];       // D x keys (transposed)
  __shared__ unsigned short Ps [8][16][40];     // per-wave P scratch 16x32

  const int tid  = threadIdx.x;
  const int wave = tid >> 5;
  const int lane = tid & 31;
  const int lh   = lane & 15;
  const int half = lane >> 4;
  const int qb   = blockIdx.x;       // t-block
  const int h    = blockIdx.y;
  const int b    = blockIdx.z;
  const int kvh  = h >> 2;
  const size_t rb = (size_t)b * 2048;  // batch row base in [B*T]

  // Load Q tile (128 x 128): 2048 async 16B chunks, 8 per thread
  for (int c = tid; c < 128 * 16; c += 256) {
    int row = c >> 4, c8 = (c & 15) * 8;
    async_load_b128(lds_offset_of(&Qs[row][c8]),
                    Q + (rb + qb * 128 + row) * 2048 + h * 128 + c8);
  }

  // per-thread K staging coordinates (fixed)
  const int krow0 = (tid * 2) >> 4;
  const int kc80  = ((tid * 2) & 15) * 8;
  const int krow1 = (tid * 2 + 1) >> 4;
  const int kc81  = ((tid * 2 + 1) & 15) * 8;
  const unsigned k_l0 = lds_offset_of(&Ks[krow0][kc80]);
  const unsigned k_l1 = lds_offset_of(&Ks[krow1][kc81]);
  // V staging: pair of adjacent key rows, transposed as packed b32
  const int vkp = (tid >> 4) * 2;
  const int vn8 = (tid & 15) * 8;

  // strength-reduced global pointers (stepped by 32 rows per key block)
  const unsigned short* kp0 = Km + (rb + krow0) * 512 + kvh * 128 + kc80;
  const unsigned short* kp1 = Km + (rb + krow1) * 512 + kvh * 128 + kc81;
  const unsigned short* vp0 = Vm + (rb + vkp)   * 512 + kvh * 128 + vn8;
  const size_t kvstep = (size_t)32 * 512;

  float m_run[8], l_run[8];
  v8f acc[8];
#pragma unroll
  for (int r = 0; r < 8; ++r) { m_run[r] = -3.0e38f; l_run[r] = 0.0f; }
#pragma unroll
  for (int d = 0; d < 8; ++d) acc[d] = zero_v8f();
  wait_async_all();
  __syncthreads();

  const int t0 = qb * 128 + wave * 16;          // wave's first q row (t index)
  const float scale = 0.08838834764831845f;     // 1/sqrt(128)
  const int kend = qb * 128 + 128;

  for (int k0 = 0; k0 < kend; k0 += 32) {
    // K tile 32 x 128: async global -> LDS (row-major: natural Q*K^T B-frags)
    async_load_b128(k_l0, kp0);
    async_load_b128(k_l1, kp1);
    // V tile 32 x 128, transposed into Vts[d][key] via v_perm_b32
    {
      uint4 d0 = *(const uint4*)vp0;           // key row vkp
      uint4 d1 = *(const uint4*)(vp0 + 512);   // key row vkp+1
      const unsigned* w0 = (const unsigned*)&d0;
      const unsigned* w1 = (const unsigned*)&d1;
#pragma unroll
      for (int wd = 0; wd < 4; ++wd) {
        unsigned lo = __builtin_amdgcn_perm(w1[wd], w0[wd], 0x05040100u);
        unsigned hi = __builtin_amdgcn_perm(w1[wd], w0[wd], 0x07060302u);
        *(unsigned*)(&Vts[vn8 + 2 * wd    ][vkp]) = lo;   // 4B aligned
        *(unsigned*)(&Vts[vn8 + 2 * wd + 1][vkp]) = hi;
      }
    }
    kp0 += kvstep; kp1 += kvstep; vp0 += kvstep;
    wait_async_all();
    __syncthreads();

    // S = Q * K^T over D=128 (4 WMMA k-steps), two 16-key subtiles
    v8f s[2];
#pragma unroll
    for (int j = 0; j < 2; ++j) s[j] = zero_v8f();
#pragma unroll
    for (int dstep = 0; dstep < 4; ++dstep) {
      const unsigned short* qr = &Qs[wave * 16 + lh][dstep * 32];
      v16bf aq = load_frag(qr + half * 8, qr + 16 + half * 8);
#pragma unroll
      for (int j = 0; j < 2; ++j) {
        const unsigned short* kr = &Ks[j * 16 + lh][dstep * 32];
        v16bf bk = load_frag(kr + half * 8, kr + 16 + half * 8);
        s[j] = wmma_bf16(aq, bk, s[j]);
      }
    }

    // scale + causal mask: element (m,n): m = r + half*8, n = lh (+ j*16 keys)
#pragma unroll
    for (int j = 0; j < 2; ++j) {
      int kidx = k0 + j * 16 + lh;
#pragma unroll
      for (int r = 0; r < 8; ++r) {
        int qidx = t0 + half * 8 + r;
        float v = s[j][r] * scale;
        s[j][r] = (kidx <= qidx) ? v : -3.0e38f;
      }
    }

    // online softmax per row (reduce across the 16 lanes owning the row)
#pragma unroll
    for (int r = 0; r < 8; ++r) {
      float mx = fmaxf(s[0][r], s[1][r]);
      mx = fmaxf(mx, __shfl_xor(mx, 1, 32));
      mx = fmaxf(mx, __shfl_xor(mx, 2, 32));
      mx = fmaxf(mx, __shfl_xor(mx, 4, 32));
      mx = fmaxf(mx, __shfl_xor(mx, 8, 32));
      float newm = fmaxf(m_run[r], mx);
      float corr = __expf(m_run[r] - newm);
      float p0 = __expf(s[0][r] - newm);
      float p1 = __expf(s[1][r] - newm);
      float rs = p0 + p1;
      rs += __shfl_xor(rs, 1, 32);
      rs += __shfl_xor(rs, 2, 32);
      rs += __shfl_xor(rs, 4, 32);
      rs += __shfl_xor(rs, 8, 32);
      l_run[r] = l_run[r] * corr + rs;
      m_run[r] = newm;
#pragma unroll
      for (int d = 0; d < 8; ++d) acc[d][r] *= corr;
      Ps[wave][half * 8 + r][lh]      = f32_to_bf16(p0);
      Ps[wave][half * 8 + r][16 + lh] = f32_to_bf16(p1);
    }

    // O += P * V  (A = P 16x32, B = V 32x16 per D-subtile from Vts)
    {
      const unsigned short* pr = &Ps[wave][lh][0];
      v16bf ap = load_frag(pr + half * 8, pr + 16 + half * 8);
#pragma unroll
      for (int d = 0; d < 8; ++d) {
        const unsigned short* vr = &Vts[d * 16 + lh][0];
        v16bf bv = load_frag(vr + half * 8, vr + 16 + half * 8);
        acc[d] = wmma_bf16(ap, bv, acc[d]);
      }
    }
    __syncthreads();
  }

  // normalize and store bf16 (row stride constant 2048 -> cheap addressing)
  unsigned short* yp = Y + (rb + t0 + half * 8) * 2048 + h * 128 + lh;
#pragma unroll
  for (int r = 0; r < 8; ++r) {
    float inv = 1.0f / l_run[r];
#pragma unroll
    for (int d = 0; d < 8; ++d)
      yp[d * 16] = f32_to_bf16(acc[d][r] * inv);
    yp += 2048;
  }
}

// ---------------------------------------------------------------------------
// Host-side orchestration
// ---------------------------------------------------------------------------
extern "C" void kernel_launch(void* const* d_in, const int* in_sizes, int n_in,
                              void* d_out, int out_size, void* d_ws, size_t ws_size,
                              hipStream_t stream) {
  const float* x  = (const float*)d_in[0];
  const float* wq = (const float*)d_in[1];
  const float* wk = (const float*)d_in[2];
  const float* wv = (const float*)d_in[3];
  const float* wc = (const float*)d_in[4];
  const float* qw = (const float*)d_in[5];
  const float* kw = (const float*)d_in[6];
  float* out = (float*)d_out;

  const size_t BT = 4096, C = 2048, KV = 512;
  char* ws = (char*)d_ws;
  size_t off = 0;
  auto alloc = [&](size_t bytes) -> char* {
    char* p = ws + off;
    off += (bytes + 255) & ~(size_t)255;
    return p;
  };
  unsigned short* xb  = (unsigned short*)alloc(BT * C * 2);
  unsigned short* wqb = (unsigned short*)alloc(C * C * 2);
  unsigned short* wkb = (unsigned short*)alloc(C * KV * 2);
  unsigned short* wvb = (unsigned short*)alloc(C * KV * 2);
  unsigned short* wcb = (unsigned short*)alloc(C * C * 2);
  float*          qf  = (float*)alloc(BT * C * 4);
  float*          kf  = (float*)alloc(BT * KV * 4);
  float*          vf  = (float*)alloc(BT * KV * 4);
  unsigned short* qb  = (unsigned short*)alloc(BT * C * 2);
  unsigned short* kb  = (unsigned short*)alloc(BT * KV * 2);
  unsigned short* vb  = (unsigned short*)alloc(BT * KV * 2);
  unsigned short* yb  = (unsigned short*)alloc(BT * C * 2);
  (void)ws_size; (void)in_sizes; (void)n_in; (void)out_size;

  // 1. bf16 conversions of activations + weights
  convert_f32_bf16_kernel<<<1024, 256, 0, stream>>>(x,  xb,  (long long)(BT * C / 4));
  convert_f32_bf16_kernel<<<1024, 256, 0, stream>>>(wq, wqb, (long long)(C * C / 4));
  convert_f32_bf16_kernel<<<512,  256, 0, stream>>>(wk, wkb, (long long)(C * KV / 4));
  convert_f32_bf16_kernel<<<512,  256, 0, stream>>>(wv, wvb, (long long)(C * KV / 4));
  convert_f32_bf16_kernel<<<1024, 256, 0, stream>>>(wc, wcb, (long long)(C * C / 4));

  // 2. QKV projections (WMMA bf16 -> f32)
  gemm_bf16_kernel<<<dim3(C / 128,  BT / 128), 256, 0, stream>>>(xb, wqb, qf, (int)BT, (int)C,  (int)C);
  gemm_bf16_kernel<<<dim3(KV / 128, BT / 128), 256, 0, stream>>>(xb, wkb, kf, (int)BT, (int)KV, (int)C);
  gemm_bf16_kernel<<<dim3(KV / 128, BT / 128), 256, 0, stream>>>(xb, wvb, vf, (int)BT, (int)KV, (int)C);

  // 3. RMSNorm + RoPE on q,k (f32 -> bf16); v: plain bf16 convert
  rmsnorm_rope_kernel<<<(unsigned)(BT * 16 / 8), 256, 0, stream>>>(qf, qb, qw, 16, (int)BT, 2048);
  rmsnorm_rope_kernel<<<(unsigned)(BT * 4  / 8), 256, 0, stream>>>(kf, kb, kw, 4,  (int)BT, 2048);
  convert_f32_bf16_kernel<<<512, 256, 0, stream>>>(vf, vb, (long long)(BT * KV / 4));

  // 4. Causal flash attention (GQA) -> y bf16
  attention_kernel<<<dim3(16, 16, 2), 256, 0, stream>>>(qb, kb, vb, yb);

  // 5. Output projection -> f32 out
  gemm_bf16_kernel<<<dim3(16, 32), 256, 0, stream>>>(yb, wcb, out, (int)BT, (int)C, (int)C);
}
